// Multiheadattention_5712306503749
// MI455X (gfx1250) — matrix operand order repaired and checked by reading, not compiled, not run on hardware
//
#include <hip/hip_runtime.h>

#define BS     4
#define SEQ    2048
#define HIDDEN 1024
#define NHEAD  16
#define HDIM   64

typedef __bf16 v16bf __attribute__((ext_vector_type(16)));
typedef float  v8f   __attribute__((ext_vector_type(8)));

union Frag { v16bf v; uint4 q[2]; };

__device__ __forceinline__ v8f wmma_bf16(v16bf a, v16bf b, v8f c) {
    return __builtin_amdgcn_wmma_f32_16x16x32_bf16(
        /*neg_a=*/false, a, /*neg_b=*/false, b,
        /*c_mod=*/(short)0, c, /*reuse_a=*/false, /*reuse_b=*/false);
}

// Async DMA: global -> LDS, 16 bytes per lane, tracked by ASYNCcnt.
__device__ __forceinline__ void async_load_b128(unsigned lds_addr, const void* gaddr) {
    asm volatile("global_load_async_to_lds_b128 %0, %1, off"
                 :: "v"(lds_addr), "v"(gaddr) : "memory");
}
__device__ __forceinline__ void wait_asynccnt0() {
    asm volatile("s_wait_asynccnt 0" ::: "memory");
}
__device__ __forceinline__ void wait_dscnt0() {
    asm volatile("s_wait_dscnt 0" ::: "memory");
}

// ---------------------------------------------------------------- convert f32 -> bf16
__global__ void f32_to_bf16_kernel(const float* __restrict__ in,
                                   __bf16* __restrict__ out, size_t n) {
    size_t i = (size_t)blockIdx.x * blockDim.x + threadIdx.x;
    size_t stride = (size_t)gridDim.x * blockDim.x;
    for (; i < n; i += stride) out[i] = (__bf16)in[i];
}

// ---------------------------------------------------------------- W[K][N] f32 -> Wt[N][K] bf16
__global__ void transpose_w_kernel(const float* __restrict__ W,
                                   __bf16* __restrict__ Wt, int K, int N) {
    size_t total = (size_t)K * N;
    size_t i = (size_t)blockIdx.x * blockDim.x + threadIdx.x;
    size_t stride = (size_t)gridDim.x * blockDim.x;
    for (; i < total; i += stride) {
        int n = (int)(i / K);
        int k = (int)(i % K);
        Wt[i] = (__bf16)W[(size_t)k * N + n];
    }
}

// ---------------------------------------------------------------- WMMA GEMM
// C[M,N] = A[M,Kd] * Bt[N,Kd]^T + bias[N].  Wave tile: 32(M) x 64(N).
// K loop manually unrolled by 2 with two named fragment sets so the double
// buffer stays in VGPRs (no dynamic indexing -> no alloca/spill) and loads
// of one set overlap WMMAs of the other.
// mode 0: bf16 row-major; mode 1: V-transpose (Vt[b][h][d][seq] bf16); mode 2: f32.
__global__ __launch_bounds__(256) void gemm_bf16_kernel(
    const __bf16* __restrict__ A, const __bf16* __restrict__ Bt,
    const float* __restrict__ bias, void* __restrict__ Cout,
    int M, int N, int Kd, int mode) {
    int lane = threadIdx.x & 31;
    int wave = threadIdx.x >> 5;
    int half = lane >> 4;
    int lm   = lane & 15;

    int tilesN = N >> 6;
    int wid = blockIdx.x * 8 + wave;
    int tm = wid / tilesN;
    int tn = wid % tilesN;
    int m0 = tm * 32;
    int n0 = tn * 64;
    if (m0 >= M) return;

    const __bf16* paRow0 = A + (size_t)(m0 + lm) * Kd;
    const __bf16* paRow1 = A + (size_t)(m0 + 16 + lm) * Kd;
    const __bf16* pbRow0 = Bt + (size_t)(n0 + lm) * Kd + 16 * half;
    const __bf16* pbRow1 = pbRow0 + (size_t)16 * Kd;
    const __bf16* pbRow2 = pbRow0 + (size_t)32 * Kd;
    const __bf16* pbRow3 = pbRow0 + (size_t)48 * Kd;

    auto loadTile = [&](Frag (&af)[2], Frag (&bf)[4], int k0) {
        const __bf16* pa0 = paRow0 + k0;
        const __bf16* pa1 = paRow1 + k0;
        af[0].q[0] = *(const uint4*)(pa0 + 16 * half);
        af[0].q[1] = *(const uint4*)(pa0 + 16 * half + 8);
        af[1].q[0] = *(const uint4*)(pa1 + 16 * half);
        af[1].q[1] = *(const uint4*)(pa1 + 16 * half + 8);
        bf[0].q[0] = *(const uint4*)(pbRow0 + k0);
        bf[0].q[1] = *(const uint4*)(pbRow0 + k0 + 8);
        bf[1].q[0] = *(const uint4*)(pbRow1 + k0);
        bf[1].q[1] = *(const uint4*)(pbRow1 + k0 + 8);
        bf[2].q[0] = *(const uint4*)(pbRow2 + k0);
        bf[2].q[1] = *(const uint4*)(pbRow2 + k0 + 8);
        bf[3].q[0] = *(const uint4*)(pbRow3 + k0);
        bf[3].q[1] = *(const uint4*)(pbRow3 + k0 + 8);
    };

    v8f acc[2][4] = {};
    auto mmas = [&](Frag (&af)[2], Frag (&bf)[4]) {
#pragma unroll
        for (int i = 0; i < 2; ++i)
#pragma unroll
            for (int j = 0; j < 4; ++j)
                acc[i][j] = wmma_bf16(af[i].v, bf[j].v, acc[i][j]);
    };

    Frag aA[2], bA[4], aB[2], bB[4];
    loadTile(aA, bA, 0);
    for (int k0 = 0; k0 + 64 <= Kd; k0 += 64) {
        loadTile(aB, bB, k0 + 32);          // prefetch odd step
        mmas(aA, bA);                        // consume even step
        if (k0 + 64 < Kd) loadTile(aA, bA, k0 + 64);   // prefetch next even step
        mmas(aB, bB);                        // consume odd step
    }

#pragma unroll
    for (int i = 0; i < 2; ++i) {
        int mi = m0 + 16 * i;
#pragma unroll
        for (int j = 0; j < 4; ++j) {
            int ng = n0 + j * 16 + lm;
            float bb = bias ? bias[ng] : 0.0f;
            if (mode == 0) {
                __bf16* C = (__bf16*)Cout;
#pragma unroll
                for (int r = 0; r < 8; ++r) {
                    int mg = mi + r + 8 * half;
                    C[(size_t)mg * N + ng] = (__bf16)(acc[i][j][r] + bb);
                }
            } else if (mode == 2) {
                float* C = (float*)Cout;
#pragma unroll
                for (int r = 0; r < 8; ++r) {
                    int mg = mi + r + 8 * half;
                    C[(size_t)mg * N + ng] = acc[i][j][r] + bb;
                }
            } else {
                __bf16* C = (__bf16*)Cout;
                int h = ng >> 6, d = ng & 63;
                int mbase = mi + 8 * half;
                int b = mbase / SEQ;
                int s = mbase % SEQ;
                union { uint4 q; __bf16 e[8]; } pk;
#pragma unroll
                for (int r = 0; r < 8; ++r) pk.e[r] = (__bf16)(acc[i][j][r] + bb);
                *(uint4*)(C + (((size_t)(b * NHEAD + h) * HDIM + d) * SEQ + s)) = pk.q;
            }
        }
    }
}

// ---------------------------------------------------------------- flash attention
// 8 waves/block share one (b,h); K/V tiles staged into LDS via async DMA,
// double-buffered so the next tile's copy overlaps this tile's compute.
__global__ __launch_bounds__(256) void attn_kernel(
    const __bf16* __restrict__ Qb, const __bf16* __restrict__ Kb,
    const __bf16* __restrict__ Vt, const int* __restrict__ mask,
    __bf16* __restrict__ X) {
    __shared__ __bf16 ldsK[2][32 * 64];   // 2 x 4KB: 32 keys x 64 d
    __shared__ __bf16 ldsV[2][64 * 32];   // 2 x 4KB: 64 d x 32 keys (from Vt)
    __shared__ __bf16 ldsP[8][16 * 40];   // per-wave P transpose, padded rows

    int t    = threadIdx.x;
    int lane = t & 31;
    int wave = t >> 5;
    int half = lane >> 4;
    int lm   = lane & 15;

    const int qchunks = SEQ / 128;
    int bh = blockIdx.x / qchunks;
    int qc = blockIdx.x % qchunks;
    int b = bh / NHEAD, h = bh % NHEAD;
    int q0 = qc * 128 + wave * 16;

    // cooperative async copy: 256 threads x 16B = 4KB per tile
    const __bf16* gKbase = Kb + (size_t)b * SEQ * HIDDEN + h * HDIM;
    const __bf16* gVbase = Vt + (size_t)(b * NHEAD + h) * HDIM * SEQ;
    unsigned lK0 = (unsigned)(size_t)(&ldsK[0][0]);
    unsigned lV0 = (unsigned)(size_t)(&ldsV[0][0]);
    int kRow = t >> 3, kChunk = t & 7;    // K tile: 32 rows x 128B
    int vRow = t >> 2, vChunk = t & 3;    // V tile: 64 rows x 64B
    auto issueTile = [&](int buf, int k0) {
        async_load_b128(lK0 + (unsigned)(buf * 4096 + kRow * 128 + kChunk * 16),
                        gKbase + (size_t)(k0 + kRow) * HIDDEN + kChunk * 8);
        async_load_b128(lV0 + (unsigned)(buf * 4096 + vRow * 64 + vChunk * 16),
                        gVbase + (size_t)vRow * SEQ + k0 + vChunk * 8);
    };

    // Q A-fragments (d 0..31, 32..63), row = q0+lm
    const __bf16* qrow = Qb + (size_t)(b * SEQ + q0 + lm) * HIDDEN + h * HDIM;
    Frag aq0, aq1;
    aq0.q[0] = *(const uint4*)(qrow + 16 * half);
    aq0.q[1] = *(const uint4*)(qrow + 16 * half + 8);
    aq1.q[0] = *(const uint4*)(qrow + 32 + 16 * half);
    aq1.q[1] = *(const uint4*)(qrow + 40 + 16 * half);

    v8f acc[4] = {};
    float mrow[8], lrow[8];
#pragma unroll
    for (int r = 0; r < 8; ++r) { mrow[r] = -1e30f; lrow[r] = 0.0f; }

    const float LOG2E = 1.44269504f;
    const int* mline = mask + (size_t)b * SEQ;
    __bf16* ldsw = &ldsP[wave][0];

    issueTile(0, 0);
    wait_asynccnt0();
    __syncthreads();

    const int NK = SEQ / 32;
    for (int it = 0; it < NK; ++it) {
        int cur = it & 1;
        int k0 = it * 32;
        if (it + 1 < NK) issueTile(cur ^ 1, k0 + 32);   // DMA next tile during compute

        // S = Q * K^T, K fragments from LDS
        const __bf16* ldsKc = &ldsK[cur][0];
        v8f s[2] = {};
#pragma unroll
        for (int c = 0; c < 2; ++c) {
            const __bf16* kr = ldsKc + (c * 16 + lm) * 64 + 16 * half;
            Frag bk0, bk1;
            bk0.q[0] = *(const uint4*)(kr);
            bk0.q[1] = *(const uint4*)(kr + 8);
            bk1.q[0] = *(const uint4*)(kr + 32);
            bk1.q[1] = *(const uint4*)(kr + 40);
            s[c] = wmma_bf16(aq0.v, bk0.v, s[c]);
            s[c] = wmma_bf16(aq1.v, bk1.v, s[c]);
        }

        float mb0 = (mline[k0 + lm]      == 0) ? -10000.0f : 0.0f;
        float mb1 = (mline[k0 + 16 + lm] == 0) ? -10000.0f : 0.0f;
#pragma unroll
        for (int r = 0; r < 8; ++r) { s[0][r] += mb0; s[1][r] += mb1; }

        float newm[8];
#pragma unroll
        for (int r = 0; r < 8; ++r) {
            float v = fmaxf(s[0][r], s[1][r]);
            v = fmaxf(v, __shfl_xor(v, 1, 16));
            v = fmaxf(v, __shfl_xor(v, 2, 16));
            v = fmaxf(v, __shfl_xor(v, 4, 16));
            v = fmaxf(v, __shfl_xor(v, 8, 16));
            newm[r] = fmaxf(mrow[r], v);
        }
#pragma unroll
        for (int r = 0; r < 8; ++r) {
            float sc = exp2f((mrow[r] - newm[r]) * LOG2E);
            float p0 = exp2f((s[0][r] - newm[r]) * LOG2E);
            float p1 = exp2f((s[1][r] - newm[r]) * LOG2E);
            float ps = p0 + p1;
            ps += __shfl_xor(ps, 1, 16);
            ps += __shfl_xor(ps, 2, 16);
            ps += __shfl_xor(ps, 4, 16);
            ps += __shfl_xor(ps, 8, 16);
            lrow[r] = lrow[r] * sc + ps;
            mrow[r] = newm[r];
#pragma unroll
            for (int j = 0; j < 4; ++j) acc[j][r] *= sc;
            int row = r + 8 * half;
            ldsw[row * 40 + lm]      = (__bf16)p0;
            ldsw[row * 40 + 16 + lm] = (__bf16)p1;
        }
        wait_dscnt0();   // P transpose: DS in-order per wave, lanes lockstep

        Frag ap;
        {
            const __bf16* pr = ldsw + lm * 40;
            ap.q[0] = *(const uint4*)(pr + 16 * half);
            ap.q[1] = *(const uint4*)(pr + 16 * half + 8);
        }

        // O += P * V, V fragments from LDS
        const __bf16* ldsVc = &ldsV[cur][0];
#pragma unroll
        for (int j = 0; j < 4; ++j) {
            const __bf16* vr = ldsVc + (j * 16 + lm) * 32 + 16 * half;
            Frag bv;
            bv.q[0] = *(const uint4*)(vr);
            bv.q[1] = *(const uint4*)(vr + 8);
            acc[j] = wmma_bf16(ap.v, bv.v, acc[j]);
        }

        wait_asynccnt0();    // my DMA for next buffer landed
        __syncthreads();     // everyone done reading cur + issued next
    }

#pragma unroll
    for (int r = 0; r < 8; ++r) {
        float inv = (lrow[r] > 0.0f) ? 1.0f / lrow[r] : 0.0f;
#pragma unroll
        for (int j = 0; j < 4; ++j) acc[j][r] *= inv;
    }
#pragma unroll
    for (int j = 0; j < 4; ++j) {
        int ng = h * HDIM + j * 16 + lm;
#pragma unroll
        for (int r = 0; r < 8; ++r) {
            int qg = q0 + r + 8 * half;
            X[(size_t)(b * SEQ + qg) * HIDDEN + ng] = (__bf16)acc[j][r];
        }
    }
}

// ---------------------------------------------------------------- launch
extern "C" void kernel_launch(void* const* d_in, const int* in_sizes, int n_in,
                              void* d_out, int out_size, void* d_ws, size_t ws_size,
                              hipStream_t stream) {
    const float* srcQ = (const float*)d_in[0];
    const float* srcK = (const float*)d_in[1];
    const float* srcV = (const float*)d_in[2];
    const int*   mask = (const int*)d_in[3];
    const float* Wq = (const float*)d_in[4];
    const float* bq = (const float*)d_in[5];
    const float* Wk = (const float*)d_in[6];
    const float* bk = (const float*)d_in[7];
    const float* Wv = (const float*)d_in[8];
    const float* bv = (const float*)d_in[9];
    const float* Wo = (const float*)d_in[10];
    const float* bo = (const float*)d_in[11];
    float* out = (float*)d_out;

    const size_t M = (size_t)BS * SEQ;           // 8192
    const size_t actElems = M * HIDDEN;
    const size_t wElems = (size_t)HIDDEN * HIDDEN;

    char* ws = (char*)d_ws;
    size_t off = 0;
    auto carve = [&](size_t bytes) -> char* {
        char* p = ws + off;
        off += (bytes + 255) & ~(size_t)255;
        return p;
    };
    __bf16* bsrcQ = (__bf16*)carve(actElems * 2);
    __bf16* bsrcK = (__bf16*)carve(actElems * 2);
    __bf16* bsrcV = (__bf16*)carve(actElems * 2);
    __bf16* WqT   = (__bf16*)carve(wElems * 2);
    __bf16* WkT   = (__bf16*)carve(wElems * 2);
    __bf16* WvT   = (__bf16*)carve(wElems * 2);
    __bf16* WoT   = (__bf16*)carve(wElems * 2);
    __bf16* Qb    = (__bf16*)carve(actElems * 2);
    __bf16* Kb    = (__bf16*)carve(actElems * 2);
    __bf16* Vt    = (__bf16*)carve(actElems * 2);
    __bf16* Xb    = (__bf16*)carve(actElems * 2);

    f32_to_bf16_kernel<<<2048, 256, 0, stream>>>(srcQ, bsrcQ, actElems);
    f32_to_bf16_kernel<<<2048, 256, 0, stream>>>(srcK, bsrcK, actElems);
    f32_to_bf16_kernel<<<2048, 256, 0, stream>>>(srcV, bsrcV, actElems);
    transpose_w_kernel<<<1024, 256, 0, stream>>>(Wq, WqT, HIDDEN, HIDDEN);
    transpose_w_kernel<<<1024, 256, 0, stream>>>(Wk, WkT, HIDDEN, HIDDEN);
    transpose_w_kernel<<<1024, 256, 0, stream>>>(Wv, WvT, HIDDEN, HIDDEN);
    transpose_w_kernel<<<1024, 256, 0, stream>>>(Wo, WoT, HIDDEN, HIDDEN);

    // waves = (M/32)*(N/64) = 4096 -> 512 blocks of 8 waves
    int gemmBlocks = (int)((M / 32) * (HIDDEN / 64) / 8);
    gemm_bf16_kernel<<<gemmBlocks, 256, 0, stream>>>(bsrcQ, WqT, bq, Qb,
                                                     (int)M, HIDDEN, HIDDEN, 0);
    gemm_bf16_kernel<<<gemmBlocks, 256, 0, stream>>>(bsrcK, WkT, bk, Kb,
                                                     (int)M, HIDDEN, HIDDEN, 0);
    gemm_bf16_kernel<<<gemmBlocks, 256, 0, stream>>>(bsrcV, WvT, bv, Vt,
                                                     (int)M, HIDDEN, HIDDEN, 1);

    attn_kernel<<<BS * NHEAD * (SEQ / 128), 256, 0, stream>>>(Qb, Kb, Vt, mask, Xb);

    gemm_bf16_kernel<<<gemmBlocks, 256, 0, stream>>>(Xb, WoT, bo, out,
                                                     (int)M, HIDDEN, HIDDEN, 2);
}
